// AttentionOnDetail_84911503442422
// MI455X (gfx1250) — compile-verified, matrix operand order checked by reading, not run on hardware
//
#include <hip/hip_runtime.h>
#include <cmath>

typedef __attribute__((ext_vector_type(16))) _Float16 v16h;
typedef __attribute__((ext_vector_type(8)))  float    v8f;

// Problem constants (match reference)
constexpr int cB  = 4;
constexpr int cT  = 4096;
constexpr int cC  = 1024;
constexpr int cNH = 16;
constexpr int cDH = 128;
constexpr int cDQ = 2048;   // d_qkv
constexpr int cDR = 256;    // d_rank
constexpr int cM  = cB * cT;            // 16384 rows
constexpr float cEPS = 1.1920929e-07f;  // finfo(float32).eps

// scan chunking
constexpr int SCH = 32;          // chunks per sequence
constexpr int CL  = cT / SCH;    // 128 timesteps per chunk

// ---------------------------------------------------------------------------
// Fragment packing: 16x32 f16 tile layout per CDNA5 ISA 7.12.2.
// Element (row r, k) -> lane = r + ((k>>3)&1)*16 ; half = (k&7) + ((k>>4)&1)*8
// Packed tile = 512 halfs, lane-major (lane*16 + half): a wave fetches one
// fragment as 32 lanes x 32 contiguous bytes.
// ---------------------------------------------------------------------------
__device__ __forceinline__ size_t frag_index(int row, int col, int Kdim) {
  int tile = (row >> 4) * (Kdim >> 5) + (col >> 5);
  int rr = row & 15, kk = col & 31;
  int lane = rr + (((kk >> 3) & 1) << 4);
  int h    = (kk & 7) + (((kk >> 4) & 1) << 3);
  return (size_t)tile * 512 + (size_t)lane * 16 + h;
}

// Generic fp32 row-major [R,K] -> packed f16 fragments (act: 0=none, 1=relu^2)
__global__ void pack_f32_to_frag(const float* __restrict__ src,
                                 _Float16* __restrict__ dst,
                                 int R, int K, int act) {
  int idx = blockIdx.x * blockDim.x + threadIdx.x;
  if (idx >= R * K) return;
  int r = idx / K, k = idx - r * K;
  float v = src[idx];
  if (act == 1) { v = fmaxf(v, 0.0f); v = v * v; }
  dst[frag_index(r, k, K)] = (_Float16)v;
}

// rmsnorm over C=1024 per row, fused with fragment packing
__global__ void rmsnorm_pack(const float* __restrict__ x,
                             _Float16* __restrict__ dst) {
  __shared__ float red[256];
  int row = blockIdx.x;
  const float* xr = x + (size_t)row * cC;
  float s = 0.0f;
  for (int i = threadIdx.x; i < cC; i += 256) { float v = xr[i]; s += v * v; }
  red[threadIdx.x] = s; __syncthreads();
  for (int off = 128; off > 0; off >>= 1) {
    if ((int)threadIdx.x < off) red[threadIdx.x] += red[threadIdx.x + off];
    __syncthreads();
  }
  float scale = rsqrtf(red[0] / (float)cC + cEPS);
  for (int k = threadIdx.x; k < cC; k += 256) {
    dst[frag_index(row, k, cC)] = (_Float16)(xr[k] * scale);
  }
}

// ---------------------------------------------------------------------------
// WMMA GEMM with double-buffered async-to-LDS staging.
// D[M,N] = A[M,K] @ W[N,K]^T, operands pre-packed in fragment tiles.
// Block = 256 thr = 8 wave32; block tile 128x128; wave tile 32x64 (2x4 WMMA).
// Per K-step: 16 x 1KB tiles (8 A + 8 B) staged to LDS via
// global_load_async_to_lds_b128 (ASYNCcnt), consumed via ds_load_b128.
// store_mode: 0 = fp32 row-major, 1 = f16 row-major, 2 = packed f16 fragments
// act:        0 = none, 1 = relu^2
// ---------------------------------------------------------------------------
__global__ void __launch_bounds__(256)
gemm_wmma(const _Float16* __restrict__ Ap, const _Float16* __restrict__ Bp,
          void* __restrict__ Dout, int Mv, int Nv, int Kv,
          int store_mode, int act) {
  __shared__ __align__(32) _Float16 sm[2][16][512];   // 2 x 16KB

  int tid  = threadIdx.x;
  int lane = tid & 31;
  int w    = tid >> 5;     // wave 0..7
  int wm   = w >> 1;       // 0..3 : 32-row strip
  int wn   = w & 1;        // 0..1 : 64-col strip
  int Kt   = Kv >> 5;
  int mtB  = blockIdx.y * 8;   // first M-tile of block
  int ntB  = blockIdx.x * 8;   // first N-tile of block

  // this wave stages LDS tiles t0,t1 (0..7 = A tiles, 8..15 = B tiles)
  int t0 = w * 2, t1 = w * 2 + 1;
  auto tileGlobal = [&](int t, int kt) -> const _Float16* {
    if (t < 8) return Ap + ((size_t)(mtB + t) * Kt + kt) * 512;
    return Bp + ((size_t)(ntB + (t - 8)) * Kt + kt) * 512;
  };
  auto issue_copy = [&](int buf, int kt) {
    const _Float16* g0 = tileGlobal(t0, kt) + lane * 16;
    const _Float16* g1 = tileGlobal(t1, kt) + lane * 16;
    unsigned l0 = (unsigned)(size_t)(&sm[buf][t0][lane * 16]);
    unsigned l1 = (unsigned)(size_t)(&sm[buf][t1][lane * 16]);
    asm volatile("global_load_async_to_lds_b128 %0, %1, off"
                 :: "v"(l0), "v"(g0) : "memory");
    asm volatile("global_load_async_to_lds_b128 %0, %1, off offset:16"
                 :: "v"(l0), "v"(g0) : "memory");
    asm volatile("global_load_async_to_lds_b128 %0, %1, off"
                 :: "v"(l1), "v"(g1) : "memory");
    asm volatile("global_load_async_to_lds_b128 %0, %1, off offset:16"
                 :: "v"(l1), "v"(g1) : "memory");
  };

  v8f acc[2][4] = {};
  issue_copy(0, 0);
  for (int kt = 0; kt < Kt; ++kt) {
    int cur = kt & 1;
    if (kt + 1 < Kt) issue_copy(cur ^ 1, kt + 1);
    if (kt + 2 < Kt) {   // global_prefetch_b8 for the K-step after next
      __builtin_prefetch(tileGlobal(t0, kt + 2), 0, 1);
      __builtin_prefetch(tileGlobal(t1, kt + 2), 0, 1);
    }
    // async loads complete in order: <=4 outstanding => current batch landed
    if (kt + 1 < Kt) asm volatile("s_wait_asynccnt 0x4" ::: "memory");
    else             asm volatile("s_wait_asynccnt 0x0" ::: "memory");
    __syncthreads();

    v16h a[2], b[4];
    for (int i = 0; i < 2; ++i)
      a[i] = ((const v16h*)sm[cur][wm * 2 + i])[lane];
    for (int j = 0; j < 4; ++j)
      b[j] = ((const v16h*)sm[cur][8 + wn * 4 + j])[lane];
    for (int i = 0; i < 2; ++i)
      for (int j = 0; j < 4; ++j)
        acc[i][j] = __builtin_amdgcn_wmma_f32_16x16x32_f16(
            false, a[i], false, b[j], (short)0, acc[i][j], false, false);

    __syncthreads();   // protect buffer before next async overwrite
  }

  // C/D layout: lane 0-15 -> M=vgpr, lane 16-31 -> M=vgpr+8; N = lane&15
  int rbase = (lane >= 16) ? 8 : 0;
  int ncol  = lane & 15;
  int m0 = blockIdx.y * 128 + wm * 32;
  int n0 = blockIdx.x * 128 + wn * 64;
  for (int i = 0; i < 2; ++i) {
    for (int j = 0; j < 4; ++j) {
      int mt = m0 + i * 16, nt = n0 + j * 16;
      for (int v = 0; v < 8; ++v) {
        int row = mt + v + rbase;
        int col = nt + ncol;
        float val = acc[i][j][v];
        if (act == 1) { val = fmaxf(val, 0.0f); val *= val; }
        if (store_mode == 0) {
          ((float*)Dout)[(size_t)row * Nv + col] = val;
        } else if (store_mode == 1) {
          ((_Float16*)Dout)[(size_t)row * Nv + col] = (_Float16)val;
        } else {
          ((_Float16*)Dout)[frag_index(row, col, Nv)] = (_Float16)val;
        }
      }
    }
  }
}

// ---------------------------------------------------------------------------
// qkv post: per (row, head): rmsnorm q,k over DH=128, rope, relu(q),
// sigmoid(k)*v. Block = 128 threads, grid = cM*cNH.
// ---------------------------------------------------------------------------
__global__ void qkv_post(const _Float16* __restrict__ qkv,
                         _Float16* __restrict__ qg,
                         _Float16* __restrict__ kvb) {
  __shared__ float red[128];
  __shared__ float qs[128];
  __shared__ float ks[128];
  int blk  = blockIdx.x;
  int row  = blk >> 4;
  int head = blk & 15;
  int t    = row & (cT - 1);
  int d    = threadIdx.x;

  const _Float16* base = qkv + (size_t)row * (3 * cDQ);
  float qv = (float)base[(head)      * cDH + d];
  float kv = (float)base[(16 + head) * cDH + d];
  float vv = (float)base[(32 + head) * cDH + d];

  red[d] = qv * qv; __syncthreads();
  for (int off = 64; off > 0; off >>= 1) {
    if (d < off) red[d] += red[d + off];
    __syncthreads();
  }
  float qscale = rsqrtf(red[0] / (float)cDH + cEPS);
  __syncthreads();
  red[d] = kv * kv; __syncthreads();
  for (int off = 64; off > 0; off >>= 1) {
    if (d < off) red[d] += red[d + off];
    __syncthreads();
  }
  float kscale = rsqrtf(red[0] / (float)cDH + cEPS);
  __syncthreads();

  qs[d] = qv * qscale;
  ks[d] = kv * kscale;
  __syncthreads();

  int j = d & 63;
  float freq = (j < 32) ? __powf(1024.0f, -(float)j / 31.0f) : 0.0f;
  float th = (float)t * freq;
  float cth = __cosf(th), sth = __sinf(th);
  float qr, kr;
  if (d < 64) {
    qr =  qs[d] * cth + qs[d + 64] * sth;
    kr =  ks[d] * cth + ks[d + 64] * sth;
  } else {
    qr = -qs[d - 64] * sth + qs[d] * cth;
    kr = -ks[d - 64] * sth + ks[d] * cth;
  }
  float qrel = fmaxf(qr, 0.0f);
  float sig  = 1.0f / (1.0f + __expf(-kr));
  size_t o = (size_t)row * cDQ + (size_t)head * cDH + d;
  qg[o]  = (_Float16)qrel;
  kvb[o] = (_Float16)(sig * vv);
}

// ---------------------------------------------------------------------------
// 3-pass chunked causal scan: y[b,t,c] = qg[b,t,c] * cumsum_t(kv[b,:,c]).
// Pass1: per-chunk sums (cB*SCH*cDQ threads). Pass2: exclusive scan over
// chunks per channel. Pass3: in-chunk prefix with carry, writes y in packed
// fragment layout (K = cDQ) for GEMM3.
// ---------------------------------------------------------------------------
__global__ void scan_pass1(const _Float16* __restrict__ kvb,
                           float* __restrict__ csum) {
  int gid = blockIdx.x * 256 + threadIdx.x;   // (b*SCH + ch)*cDQ + c
  int c  = gid & (cDQ - 1);
  int r  = gid >> 11;
  int ch = r & (SCH - 1);
  int b  = r >> 5;
  float s = 0.0f;
  size_t base = ((size_t)b * cT + (size_t)ch * CL) * cDQ + c;
  for (int t = 0; t < CL; ++t) s += (float)kvb[base + (size_t)t * cDQ];
  csum[gid] = s;
}

__global__ void scan_pass2(float* __restrict__ csum) {
  int gid = blockIdx.x * 256 + threadIdx.x;   // b*cDQ + c
  int c = gid & (cDQ - 1);
  int b = gid >> 11;
  float acc = 0.0f;
  for (int ch = 0; ch < SCH; ++ch) {
    size_t o = ((size_t)(b * SCH + ch)) * cDQ + c;
    float v = csum[o];
    csum[o] = acc;          // make exclusive
    acc += v;
  }
}

__global__ void scan_pass3(const _Float16* __restrict__ qg,
                           const _Float16* __restrict__ kvb,
                           const float* __restrict__ csum,
                           _Float16* __restrict__ yp) {
  int gid = blockIdx.x * 256 + threadIdx.x;
  int c  = gid & (cDQ - 1);
  int r  = gid >> 11;
  int ch = r & (SCH - 1);
  int b  = r >> 5;
  float acc = csum[gid];
  int row0 = b * cT + ch * CL;
  for (int t = 0; t < CL; ++t) {
    int row = row0 + t;
    size_t o = (size_t)row * cDQ + c;
    acc += (float)kvb[o];
    float y = (float)qg[o] * acc;
    yp[frag_index(row, c, cDQ)] = (_Float16)y;
  }
}

// out = x + u * silu(g), uv = [u | g] in fp32
__global__ void final_gate(const float* __restrict__ x,
                           const float* __restrict__ uv,
                           float* __restrict__ out) {
  int idx = blockIdx.x * blockDim.x + threadIdx.x;
  if (idx >= cM * cC) return;
  int row = idx >> 10;
  int cc  = idx & (cC - 1);
  float u = uv[(size_t)row * cDQ + cc];
  float g = uv[(size_t)row * cDQ + cC + cc];
  float sg = g / (1.0f + __expf(-g));
  out[idx] = x[idx] + u * sg;
}

// ---------------------------------------------------------------------------
extern "C" void kernel_launch(void* const* d_in, const int* in_sizes, int n_in,
                              void* d_out, int out_size, void* d_ws, size_t ws_size,
                              hipStream_t stream) {
  const float* x   = (const float*)d_in[0];
  const float* w1  = (const float*)d_in[1];   // qkv_w1   [DR, C]
  const float* w2  = (const float*)d_in[2];   // qkv_w2   [3*DQ, DR]
  const float* cw1 = (const float*)d_in[3];   // cproj_w1 [DR, DQ]
  const float* cw2 = (const float*)d_in[4];   // cproj_w2 [2*C, DR]

  char* ws = (char*)d_ws;
  size_t off = 0;
  auto alloc = [&](size_t bytes) -> char* {
    char* p = ws + off;
    off += (bytes + 255) & ~(size_t)255;
    return p;
  };

  _Float16* w1p  = (_Float16*)alloc((size_t)cDR * cC  * 2);       // 0.5 MB
  _Float16* w2p  = (_Float16*)alloc((size_t)3 * cDQ * cDR * 2);   // 3 MB
  _Float16* cw1p = (_Float16*)alloc((size_t)cDR * cDQ * 2);       // 1 MB
  _Float16* cw2p = (_Float16*)alloc((size_t)2 * cC * cDR * 2);    // 1 MB
  float*    csum = (float*)alloc((size_t)cB * SCH * cDQ * 4);     // 1 MB
  _Float16* xnp  = (_Float16*)alloc((size_t)cM * cC * 2);         // 32 MB
  _Float16* hp   = (_Float16*)alloc((size_t)cM * cDR * 2);        // 8 MB
  _Float16* t1p  = (_Float16*)alloc((size_t)cM * cDR * 2);        // 8 MB
  _Float16* qg   = (_Float16*)alloc((size_t)cM * cDQ * 2);        // 64 MB
  _Float16* kvb  = (_Float16*)alloc((size_t)cM * cDQ * 2);        // 64 MB
  char*     big  = alloc((size_t)cM * 3 * cDQ * 2);               // 192 MB
  _Float16* qkvh = (_Float16*)big;                       // qkv f16 (dies after qkv_post)
  _Float16* yp   = (_Float16*)big;                       // reuse: y packed (64 MB)
  float*    uv   = (float*)(big + (size_t)cM * cDQ * 2); // reuse: uv fp32 (128 MB)

  // 1) pack weights into WMMA fragment layout
  pack_f32_to_frag<<<(cDR * cC + 255) / 256, 256, 0, stream>>>(w1, w1p, cDR, cC, 0);
  pack_f32_to_frag<<<(3 * cDQ * cDR + 255) / 256, 256, 0, stream>>>(w2, w2p, 3 * cDQ, cDR, 0);
  pack_f32_to_frag<<<(cDR * cDQ + 255) / 256, 256, 0, stream>>>(cw1, cw1p, cDR, cDQ, 0);
  pack_f32_to_frag<<<(2 * cC * cDR + 255) / 256, 256, 0, stream>>>(cw2, cw2p, 2 * cC, cDR, 0);

  // 2) rmsnorm(x) + pack
  rmsnorm_pack<<<cM, 256, 0, stream>>>(x, xnp);

  // 3) GEMM1: h = relu(rmsnorm(x) @ w1^T)^2   -> packed f16 (K=256)
  gemm_wmma<<<dim3(cDR / 128, cM / 128), 256, 0, stream>>>(
      xnp, w1p, hp, cM, cDR, cC, /*store*/2, /*act*/1);

  // 4) GEMM2: qkv = h @ w2^T                  -> f16 row-major [M, 6144]
  gemm_wmma<<<dim3(3 * cDQ / 128, cM / 128), 256, 0, stream>>>(
      hp, w2p, qkvh, cM, 3 * cDQ, cDR, /*store*/1, /*act*/0);

  // 5) head rmsnorm + rope + relu(q), sigmoid(k)*v
  qkv_post<<<cM * cNH, 128, 0, stream>>>(qkvh, qg, kvb);

  // 6) chunked causal scan -> y in packed layout (K=2048)
  scan_pass1<<<cB * SCH * cDQ / 256, 256, 0, stream>>>(kvb, csum);
  scan_pass2<<<cB * cDQ / 256, 256, 0, stream>>>(csum);
  scan_pass3<<<cB * SCH * cDQ / 256, 256, 0, stream>>>(qg, kvb, csum, yp);

  // 7) GEMM3: t1 = y @ cw1^T                  -> packed f16 (K=256)
  gemm_wmma<<<dim3(cDR / 128, cM / 128), 256, 0, stream>>>(
      yp, cw1p, t1p, cM, cDR, cDQ, /*store*/2, /*act*/0);

  // 8) GEMM4: uv = t1 @ cw2^T                 -> fp32 row-major [M, 2048]
  gemm_wmma<<<dim3(cDQ / 128, cM / 128), 256, 0, stream>>>(
      t1p, cw2p, uv, cM, cDQ, cDR, /*store*/0, /*act*/0);

  // 9) gated residual
  final_gate<<<(cM * cC + 255) / 256, 256, 0, stream>>>(x, uv, (float*)d_out);
}
